// D_Attention_82377472738015
// MI455X (gfx1250) — matrix operand order repaired and checked by reading, not compiled, not run on hardware
//
#include <hip/hip_runtime.h>
#include <math.h>

// ---------------- problem constants ----------------
#define BATCH 8
#define SEQ   1024
#define DIMC  512
#define HEADS 8
#define HDIM  64
#define SIZEC 512      // HEADS*HDIM
#define DFFC  2048
#define MROWS (BATCH * SEQ)   // 8192
#define NEGINF (-1.0e30f)

typedef __attribute__((ext_vector_type(16))) _Float16 v16h;
typedef __attribute__((ext_vector_type(8)))  float    v8f;

// ---------------- WMMA fragment helpers (CDNA5 16x16x32 f16) ----------------
// A (16x32, row-major src): lane l -> row m=l&15; halves 0..7 -> K = kb..kb+7,
// halves 8..15 -> K = 16+kb..16+kb+7, kb = (l>>4)*8.   (ISA 7.12.2)
__device__ __forceinline__ v16h load_a_frag_f32(const float* __restrict__ A,
                                                int lda, float scale) {
  const int l  = threadIdx.x & 31;
  const int m  = l & 15;
  const int kb = (l >> 4) << 3;
  const float* p0 = A + (size_t)m * lda + kb;
  const float* p1 = p0 + 16;
  v16h f;
#pragma unroll
  for (int i = 0; i < 8; ++i) f[i]     = (_Float16)(p0[i] * scale);
#pragma unroll
  for (int i = 0; i < 8; ++i) f[8 + i] = (_Float16)(p1[i] * scale);
  return f;
}

__device__ __forceinline__ v16h load_a_frag_h(const _Float16* __restrict__ A, int lda) {
  const int l  = threadIdx.x & 31;
  const int m  = l & 15;
  const int kb = (l >> 4) << 3;
  const _Float16* p0 = A + (size_t)m * lda + kb;
  const _Float16* p1 = p0 + 16;
  v16h f;
#pragma unroll
  for (int i = 0; i < 8; ++i) f[i]     = p0[i];
#pragma unroll
  for (int i = 0; i < 8; ++i) f[8 + i] = p1[i];
  return f;
}

// B (32x16): lane l -> column n=l&15; halves 0..7 -> K = kb..kb+7,
// halves 8..15 -> K = 16+kb..16+kb+7, kb = (l>>4)*8 (same order as A).
// Source supplied "transposed" (row n holds the K-run).
__device__ __forceinline__ v16h load_bt_frag_h(const _Float16* __restrict__ BT, int ld) {
  const int l  = threadIdx.x & 31;
  const int n  = l & 15;
  const int kb = (l >> 4) << 3;
  const _Float16* p = BT + (size_t)n * ld + kb;
  v16h f;
#pragma unroll
  for (int i = 0; i < 8; ++i) f[i] = p[i];
#pragma unroll
  for (int i = 0; i < 8; ++i) f[8 + i] = p[16 + i];
  return f;
}

__device__ __forceinline__ v8f wmma_f16(v16h a, v16h b, v8f c) {
  return __builtin_amdgcn_wmma_f32_16x16x32_f16(false, a, false, b,
                                                (short)0, c, false, false);
}

// ---------------- generic GEMM: C = A(f32)[M,K] x W + bias (+res) (relu?) ----
// BT is the f16 *transposed* weight: BT[n*K + k] = W[k*N + n].
// Wave tile 32x64, 4 waves/block -> 128x64 block tile. M%128==0, N%64==0, K%32==0.
// Epilogue variants are compile-time to keep the tail branch-free.
template <bool HAS_RES, bool RELU>
__global__ __launch_bounds__(128) void gemm_wmma_kernel(
    const float* __restrict__ A, const _Float16* __restrict__ BT,
    const float* __restrict__ bias, const float* __restrict__ res,
    float* __restrict__ C, int M, int N, int K) {
  const int wave = threadIdx.x >> 5;
  const int m0 = blockIdx.y * 128 + wave * 32;
  const int n0 = blockIdx.x * 64;
  v8f acc[2][4] = {};
  for (int k0 = 0; k0 < K; k0 += 32) {
    v16h a0 = load_a_frag_f32(A + (size_t)m0 * K + k0, K, 1.0f);
    v16h a1 = load_a_frag_f32(A + (size_t)(m0 + 16) * K + k0, K, 1.0f);
#pragma unroll
    for (int j = 0; j < 4; ++j) {
      v16h b = load_bt_frag_h(BT + (size_t)(n0 + 16 * j) * K + k0, K);
      acc[0][j] = wmma_f16(a0, b, acc[0][j]);
      acc[1][j] = wmma_f16(a1, b, acc[1][j]);
    }
  }
  const int l = threadIdx.x & 31;
  const int n = l & 15;
  const int mb = (l >> 4) << 3;
#pragma unroll
  for (int j = 0; j < 4; ++j) {
    const int col = n0 + 16 * j + n;
    const float bv = bias[col];
#pragma unroll
    for (int i = 0; i < 2; ++i) {
#pragma unroll
      for (int r = 0; r < 8; ++r) {
        const int row = m0 + 16 * i + mb + r;
        float v = acc[i][j][r] + bv;
        if constexpr (HAS_RES) v += res[(size_t)row * N + col];
        if constexpr (RELU)    v = fmaxf(v, 0.0f);
        C[(size_t)row * N + col] = v;
      }
    }
  }
}

// ---------------- flash attention (1 wave = one (b,h,16-query) tile) --------
// Q: f32 [B,Sq,SIZEC]; Kh: f16 [B,Skv,SIZEC]; Vt: f16 [B,H,HDIM,Skv]; O: f32 [B,Sq,SIZEC]
#define FA_STRIDE 40
__global__ __launch_bounds__(128) void flash_attn_kernel(
    const float* __restrict__ Q, const _Float16* __restrict__ Kh,
    const _Float16* __restrict__ Vt, float* __restrict__ O,
    int Sq, int Skv, int causal, float scale) {
  __shared__ _Float16 plds[4][16 * FA_STRIDE];
  const int wib = threadIdx.x >> 5;
  const int qtiles = Sq >> 4;
  const int w  = blockIdx.x * 4 + wib;
  const int qt = w % qtiles;
  const int h  = (w / qtiles) % HEADS;
  const int b  = w / (qtiles * HEADS);
  const int q0 = qt << 4;
  const int l  = threadIdx.x & 31;
  const int nn = l & 15;
  const int mb = (l >> 4) << 3;
  _Float16* pt = plds[wib];

  const float* qbase = Q + (size_t)(b * Sq + q0) * SIZEC + h * HDIM;
  const v16h qf0 = load_a_frag_f32(qbase,       SIZEC, scale);
  const v16h qf1 = load_a_frag_f32(qbase + 32,  SIZEC, scale);

  const _Float16* kbase = Kh + (size_t)b * Skv * SIZEC + h * HDIM;          // key-major
  const _Float16* vbase = Vt + (size_t)(b * HEADS + h) * HDIM * Skv;        // d-major

  v8f o[4] = {};
  float rm[8], rl[8];
#pragma unroll
  for (int r = 0; r < 8; ++r) { rm[r] = NEGINF; rl[r] = 0.0f; }

  const int nch = causal ? ((q0 + 16 + 31) >> 5) : (Skv >> 5);
  for (int c = 0; c < nch; ++c) {
    const int k0 = c << 5;
    v8f s[2];
#pragma unroll
    for (int t = 0; t < 2; ++t) {
      const int kcol = k0 + 16 * t;
      v8f z = {};
      v16h kf0 = load_bt_frag_h(kbase + (size_t)kcol * SIZEC,      SIZEC);  // d 0..31
      v16h kf1 = load_bt_frag_h(kbase + (size_t)kcol * SIZEC + 32, SIZEC);  // d 32..63
      z = wmma_f16(qf0, kf0, z);
      z = wmma_f16(qf1, kf1, z);
      if (causal) {
        const int key = kcol + nn;
#pragma unroll
        for (int r = 0; r < 8; ++r)
          if (key > q0 + mb + r) z[r] = NEGINF;
      }
      s[t] = z;
    }
    // online softmax over this 32-key chunk (rows live on 16-lane half-waves)
#pragma unroll
    for (int r = 0; r < 8; ++r) {
      float cm = fmaxf(s[0][r], s[1][r]);
      for (int mk = 8; mk >= 1; mk >>= 1) cm = fmaxf(cm, __shfl_xor(cm, mk, 16));
      const float mnew = fmaxf(rm[r], cm);
      const float p0 = __expf(s[0][r] - mnew);
      const float p1 = __expf(s[1][r] - mnew);
      pt[(mb + r) * FA_STRIDE + nn]      = (_Float16)p0;
      pt[(mb + r) * FA_STRIDE + 16 + nn] = (_Float16)p1;
      float cs = p0 + p1;
      for (int mk = 8; mk >= 1; mk >>= 1) cs += __shfl_xor(cs, mk, 16);
      const float alpha = __expf(rm[r] - mnew);
      rl[r] = rl[r] * alpha + cs;
      rm[r] = mnew;
#pragma unroll
      for (int j = 0; j < 4; ++j) o[j][r] *= alpha;
    }
    // wave-local LDS handoff: C-layout P -> A-layout fragment
    asm volatile("s_wait_dscnt 0" ::: "memory");
    const v16h pf = load_a_frag_h(pt, FA_STRIDE);
#pragma unroll
    for (int j = 0; j < 4; ++j) {
      v16h vf = load_bt_frag_h(vbase + (size_t)(16 * j) * Skv + k0, Skv);
      o[j] = wmma_f16(pf, vf, o[j]);
    }
    asm volatile("s_wait_dscnt 0" ::: "memory");  // WAR guard vs next-iter stores
  }

  float* obase = O + (size_t)(b * Sq + q0) * SIZEC + h * HDIM;
  float rinv[8];
#pragma unroll
  for (int r = 0; r < 8; ++r) rinv[r] = __frcp_rn(rl[r]);
#pragma unroll
  for (int j = 0; j < 4; ++j)
#pragma unroll
    for (int r = 0; r < 8; ++r)
      obase[(size_t)(mb + r) * SIZEC + 16 * j + nn] = o[j][r] * rinv[r];
}

// ---------------- small data-movement kernels ----------------
__global__ void wt_transpose_kernel(const float* __restrict__ W,
                                    _Float16* __restrict__ WT, int K, int N) {
  const size_t idx = (size_t)blockIdx.x * blockDim.x + threadIdx.x;  // over N*K
  const int k = (int)(idx % K);
  const int n = (int)(idx / K);
  WT[idx] = (_Float16)W[(size_t)k * N + n];
}

__global__ void cvt_f16_kernel(const float* __restrict__ in,
                               _Float16* __restrict__ out) {
  const size_t idx = (size_t)blockIdx.x * blockDim.x + threadIdx.x;
  out[idx] = (_Float16)in[idx];
}

// Vt[b,h,d,s] = (f16) V[b,s,h*HDIM+d]
__global__ void v_transpose_kernel(const float* __restrict__ V,
                                   _Float16* __restrict__ Vt, int S) {
  const size_t idx = (size_t)blockIdx.x * blockDim.x + threadIdx.x;  // B*H*HDIM*S
  const int s = (int)(idx % S);
  size_t t = idx / S;
  const int d = (int)(t % HDIM); t /= HDIM;
  const int h = (int)(t % HEADS);
  const int b = (int)(t / HEADS);
  Vt[idx] = (_Float16)V[(size_t)(b * S + s) * SIZEC + h * HDIM + d];
}

// in-place layernorm over rows of length 512 (biased variance, eps 1e-5)
__global__ __launch_bounds__(128) void layernorm_kernel(
    float* __restrict__ X, const float* __restrict__ g,
    const float* __restrict__ bb, int N) {
  __shared__ float ssum[4], ssq[4], sstat[2];
  float* x = X + (size_t)blockIdx.x * N;
  float v[4];
  float s = 0.0f, sq = 0.0f;
#pragma unroll
  for (int i = 0; i < 4; ++i) {
    v[i] = x[threadIdx.x + 128 * i];
    s += v[i]; sq += v[i] * v[i];
  }
  for (int mk = 16; mk >= 1; mk >>= 1) {
    s  += __shfl_xor(s,  mk, 32);
    sq += __shfl_xor(sq, mk, 32);
  }
  if ((threadIdx.x & 31) == 0) { ssum[threadIdx.x >> 5] = s; ssq[threadIdx.x >> 5] = sq; }
  __syncthreads();
  if (threadIdx.x == 0) {
    const float ts = ssum[0] + ssum[1] + ssum[2] + ssum[3];
    const float tq = ssq[0] + ssq[1] + ssq[2] + ssq[3];
    const float mu = ts / N;
    sstat[0] = mu;
    sstat[1] = rsqrtf(tq / N - mu * mu + 1e-5f);
  }
  __syncthreads();
  const float mu = sstat[0], rstd = sstat[1];
#pragma unroll
  for (int i = 0; i < 4; ++i) {
    const int cix = threadIdx.x + 128 * i;
    x[cix] = (v[i] - mu) * rstd * g[cix] + bb[cix];
  }
}

// ---------------- host orchestration ----------------
static inline void run_gemm(const float* A, const _Float16* BT, const float* bias,
                            const float* res, float* C, int M, int N, int K,
                            int relu, hipStream_t s) {
  dim3 grid(N / 64, M / 128);
  if (res) {
    gemm_wmma_kernel<true, false><<<grid, 128, 0, s>>>(A, BT, bias, res, C, M, N, K);
  } else if (relu) {
    gemm_wmma_kernel<false, true><<<grid, 128, 0, s>>>(A, BT, bias, res, C, M, N, K);
  } else {
    gemm_wmma_kernel<false, false><<<grid, 128, 0, s>>>(A, BT, bias, res, C, M, N, K);
  }
}

extern "C" void kernel_launch(void* const* d_in, const int* in_sizes, int n_in,
                              void* d_out, int out_size, void* d_ws, size_t ws_size,
                              hipStream_t stream) {
  (void)in_sizes; (void)n_in; (void)out_size; (void)ws_size;
  const float* x       = (const float*)d_in[0];
  const float* feature = (const float*)d_in[1];
  const float* wq  = (const float*)d_in[2];  const float* bq  = (const float*)d_in[3];
  const float* wk  = (const float*)d_in[4];  const float* bk  = (const float*)d_in[5];
  const float* wv  = (const float*)d_in[6];  const float* bv  = (const float*)d_in[7];
  const float* wo  = (const float*)d_in[8];  const float* bo  = (const float*)d_in[9];
  const float* ln1_g = (const float*)d_in[10]; const float* ln1_b = (const float*)d_in[11];
  const float* wqc = (const float*)d_in[12]; const float* bqc = (const float*)d_in[13];
  const float* wkc = (const float*)d_in[14]; const float* bkc = (const float*)d_in[15];
  const float* wvc = (const float*)d_in[16]; const float* bvc = (const float*)d_in[17];
  const float* woc = (const float*)d_in[18]; const float* boc = (const float*)d_in[19];
  const float* ln2_g = (const float*)d_in[20]; const float* ln2_b = (const float*)d_in[21];
  const float* w1  = (const float*)d_in[22]; const float* b1  = (const float*)d_in[23];
  const float* w2  = (const float*)d_in[24]; const float* b2  = (const float*)d_in[25];
  const float* lnf_g = (const float*)d_in[26]; const float* lnf_b = (const float*)d_in[27];
  float* out = (float*)d_out;

  char* ws = (char*)d_ws;
  float*    fQ  = (float*)   (ws + (0ull  << 20));
  float*    fK  = (float*)   (ws + (16ull << 20));
  _Float16* fKh = (_Float16*)(ws + (32ull << 20));
  _Float16* fVt = (_Float16*)(ws + (40ull << 20));
  float*    fV  = (float*)   (ws + (48ull << 20));
  float*    fO  = (float*)   (ws + (64ull << 20));
  float*    fH1 = (float*)   (ws + (80ull << 20));
  float*    fH2 = (float*)   (ws + (96ull << 20));
  _Float16* wTb = (_Float16*)(ws + (112ull << 20));
  float*    ff1 = (float*)   (ws + (0ull  << 20));   // reuses 0..64MB after attention

  const int WSQ = DIMC * SIZEC;   // 512*512
  _Float16* wqT  = wTb;            _Float16* wkT  = wqT  + WSQ;
  _Float16* wvT  = wkT + WSQ;      _Float16* woT  = wvT  + WSQ;
  _Float16* wqcT = woT + WSQ;      _Float16* wkcT = wqcT + WSQ;
  _Float16* wvcT = wkcT + WSQ;     _Float16* wocT = wvcT + WSQ;
  _Float16* w1T  = wocT + WSQ;                       // 2048*512
  _Float16* w2T  = w1T + DIMC * DFFC;                // 512*2048

  const float scale = 1.0f / sqrtf((float)DIMC);

  // weight transpose+convert (tiny, once per launch)
  const int TB = 256;
  wt_transpose_kernel<<<WSQ / TB, TB, 0, stream>>>(wq,  wqT,  DIMC, SIZEC);
  wt_transpose_kernel<<<WSQ / TB, TB, 0, stream>>>(wk,  wkT,  DIMC, SIZEC);
  wt_transpose_kernel<<<WSQ / TB, TB, 0, stream>>>(wv,  wvT,  DIMC, SIZEC);
  wt_transpose_kernel<<<WSQ / TB, TB, 0, stream>>>(wo,  woT,  SIZEC, DIMC);
  wt_transpose_kernel<<<WSQ / TB, TB, 0, stream>>>(wqc, wqcT, DIMC, SIZEC);
  wt_transpose_kernel<<<WSQ / TB, TB, 0, stream>>>(wkc, wkcT, DIMC, SIZEC);
  wt_transpose_kernel<<<WSQ / TB, TB, 0, stream>>>(wvc, wvcT, DIMC, SIZEC);
  wt_transpose_kernel<<<WSQ / TB, TB, 0, stream>>>(woc, wocT, SIZEC, DIMC);
  wt_transpose_kernel<<<(DIMC * DFFC) / TB, TB, 0, stream>>>(w1, w1T, DIMC, DFFC);
  wt_transpose_kernel<<<(DIMC * DFFC) / TB, TB, 0, stream>>>(w2, w2T, DFFC, DIMC);

  const size_t actN = (size_t)MROWS * SIZEC;           // 8192*512
  const dim3 fgrid((BATCH * HEADS * (SEQ / 16)) / 4);  // 1024 blocks, 4 waves each

  // ---- masked self-attention ----
  run_gemm(x, wqT, bq, nullptr, fQ, MROWS, SIZEC, DIMC, 0, stream);
  run_gemm(x, wkT, bk, nullptr, fK, MROWS, SIZEC, DIMC, 0, stream);
  run_gemm(x, wvT, bv, nullptr, fV, MROWS, SIZEC, DIMC, 0, stream);
  cvt_f16_kernel<<<actN / TB, TB, 0, stream>>>(fK, fKh);
  v_transpose_kernel<<<actN / TB, TB, 0, stream>>>(fV, fVt, SEQ);
  flash_attn_kernel<<<fgrid, 128, 0, stream>>>(fQ, fKh, fVt, fO, SEQ, SEQ, 1, scale);
  run_gemm(fO, woT, bo, x, fH1, MROWS, DIMC, SIZEC, 0, stream);
  layernorm_kernel<<<MROWS, 128, 0, stream>>>(fH1, ln1_g, ln1_b, DIMC);

  // ---- cross-attention over feature ----
  run_gemm(fH1,     wqcT, bqc, nullptr, fQ, MROWS, SIZEC, DIMC, 0, stream);
  run_gemm(feature, wkcT, bkc, nullptr, fK, MROWS, SIZEC, DIMC, 0, stream);
  run_gemm(feature, wvcT, bvc, nullptr, fV, MROWS, SIZEC, DIMC, 0, stream);
  cvt_f16_kernel<<<actN / TB, TB, 0, stream>>>(fK, fKh);
  v_transpose_kernel<<<actN / TB, TB, 0, stream>>>(fV, fVt, SEQ);
  flash_attn_kernel<<<fgrid, 128, 0, stream>>>(fQ, fKh, fVt, fO, SEQ, SEQ, 0, scale);
  run_gemm(fO, wocT, boc, fH1, fH2, MROWS, DIMC, SIZEC, 0, stream);
  layernorm_kernel<<<MROWS, 128, 0, stream>>>(fH2, ln2_g, ln2_b, DIMC);

  // ---- FFN + residual + final LN ----
  run_gemm(fH2, w1T, b1, nullptr, ff1, MROWS, DFFC, DIMC, 1, stream);   // relu
  run_gemm(ff1, w2T, b2, fH2, out, MROWS, DIMC, DFFC, 0, stream);
  layernorm_kernel<<<MROWS, 128, 0, stream>>>(out, lnf_g, lnf_b, DIMC);

  // ---- second tuple output: pass-through feature ----
  hipMemcpyAsync(out + actN, feature, actN * sizeof(float),
                 hipMemcpyDeviceToDevice, stream);
}
